// WaveNet_21208548508141
// MI455X (gfx1250) — matrix operand order repaired and checked by reading, not compiled, not run on hardware
//
#include <hip/hip_runtime.h>
#include <hip/hip_bf16.h>

// ---------------------------------------------------------------------------
// Fused WaveNet forward for MI455X (gfx1250), wave32, WMMA bf16 16x16x32.
// One kernel: init 1x1 convs -> 15 gated dilated layers -> head -> softmax.
// All intermediates live in LDS; HBM traffic = x in (0.5MB) + probs out (128MB).
// ---------------------------------------------------------------------------

typedef __attribute__((ext_vector_type(16))) __bf16 v16bf;
typedef __attribute__((ext_vector_type(8)))  __bf16 v8bf;
typedef __attribute__((ext_vector_type(8)))  float  v8f;
typedef __attribute__((ext_vector_type(4)))  float  v4f;

#define T_LEN   16384
#define BATCH_  8
#define NLAYERS 15
#define TS      512                 // output time-steps per workgroup
#define HALO    96                  // >= total receptive field 3*(1+2+4+8+16)=93
#define PADT    (TS + HALO)         // 608 columns processed per workgroup
#define BLOCK   256                 // 8 wave32 waves
#define WAVES   8
#define NT_PAD  (PADT / 16)         // 38 WMMA column tiles
#define NT_OUT  (TS / 16)           // 32 output column tiles

static_assert(PADT % 16 == 0, "PADT must be a multiple of 16");
static_assert(T_LEN % TS == 0, "T must tile evenly");

// ---- LDS layout (bytes) ----------------------------------------------------
// h double buffer + skip accumulator, f32, layout [t][32ch] (128B per column)
#define H_BYTES      (PADT * 32 * 4)
#define OFF_H0       0
#define OFF_H1       (OFF_H0 + H_BYTES)
#define OFF_SK       (OFF_H1 + H_BYTES)
// per-layer staged weights, bf16 row-major [out][K]
#define OFF_WF       (OFF_SK + H_BYTES)          // 32x64 bf16 = 4096B
#define OFF_WG       (OFF_WF + 4096)             // 32x64 bf16
#define OFF_WS       (OFF_WG + 4096)             // 32x32 bf16 = 2048B
#define OFF_WR       (OFF_WS + 2048)             // 32x32 bf16
#define OFF_BF       (OFF_WR + 2048)             // 32 f32
#define OFF_BG       (OFF_BF + 128)
#define OFF_BS       (OFF_BG + 128)
#define OFF_BR       (OFF_BS + 128)
// head weights (staged once)
#define OFF_A1W      (OFF_BR + 128)              // 32x32 bf16
#define OFF_A2W      (OFF_A1W + 2048)            // 256x32 bf16 = 16384B
#define OFF_A1B      (OFF_A2W + 16384)           // 32 f32
#define OFF_A2B      (OFF_A1B + 128)             // 256 f32
// per-wave transpose pad: 16 cols x 32 ch bf16 = 1KB each
#define OFF_TR       (OFF_A2B + 1024)
#define SMEM_BYTES   (OFF_TR + WAVES * 1024)     // = 274048 bytes < 320KB

union ABU { v16bf v; v8bf h[2]; };

__device__ __forceinline__ v8f wmma_bf16(v16bf a, v16bf b, v8f c) {
  return __builtin_amdgcn_wmma_f32_16x16x32_bf16(false, a, false, b,
                                                 (short)0, c, false, false);
}

// A operand tile 16x32 bf16 from LDS row-major weights.
// lane<16 holds M=lane, K = 0..15; lane>=16 holds K = 16..31 (matches B operand).
__device__ __forceinline__ v16bf load_a(const __bf16* w, int rowBytes,
                                        int m0, int kcBytes, int m, int kh) {
  const char* base = (const char*)w + (m0 + m) * rowBytes + kcBytes + kh * 32;
  ABU u;
  u.h[0] = *(const v8bf*)(base);
  u.h[1] = *(const v8bf*)(base + 16);
  return u.v;
}

// B operand tile 32x16 from f32 h stream [t][32ch]: lane<16 -> ch 0..15 of col,
// lane>=16 -> ch 16..31 of col; convert f32 -> bf16 on the fly.
__device__ __forceinline__ v16bf load_h_b(const float* h, int col, int kh) {
  const v4f* p = (const v4f*)(h + col * 32 + kh * 16);
  v4f q0 = p[0], q1 = p[1], q2 = p[2], q3 = p[3];
  v16bf r;
#pragma unroll
  for (int i = 0; i < 4; ++i) {
    r[i]      = (__bf16)q0[i];
    r[4 + i]  = (__bf16)q1[i];
    r[8 + i]  = (__bf16)q2[i];
    r[12 + i] = (__bf16)q3[i];
  }
  return r;
}

__device__ __forceinline__ v16bf load_skip_relu_b(const float* sk, int col, int kh) {
  const v4f* p = (const v4f*)(sk + col * 32 + kh * 16);
  v4f q0 = p[0], q1 = p[1], q2 = p[2], q3 = p[3];
  v16bf r;
#pragma unroll
  for (int i = 0; i < 4; ++i) {
    r[i]      = (__bf16)fmaxf(q0[i], 0.f);
    r[4 + i]  = (__bf16)fmaxf(q1[i], 0.f);
    r[8 + i]  = (__bf16)fmaxf(q2[i], 0.f);
    r[12 + i] = (__bf16)fmaxf(q3[i], 0.f);
  }
  return r;
}

// Bias as accumulator init: elem r of v8f = bias[m0 + 8*kh + r].
__device__ __forceinline__ v8f load_bias_tile(const float* b, int m0, int kh) {
  const v4f* p = (const v4f*)(b + m0 + kh * 8);
  v4f a = p[0], c = p[1];
  v8f r;
#pragma unroll
  for (int i = 0; i < 4; ++i) { r[i] = a[i]; r[4 + i] = c[i]; }
  return r;
}

// Store a D-tile (v8f) into the per-wave transpose pad as bf16 [col][ch].
__device__ __forceinline__ void store_tr(__bf16* scr, int ncol, int tile, int kh, v8f v) {
  v8bf o;
#pragma unroll
  for (int i = 0; i < 8; ++i) o[i] = (__bf16)v[i];
  *(v8bf*)(scr + ncol * 32 + tile * 16 + kh * 8) = o;
}

__device__ __forceinline__ v16bf load_tr_b(const __bf16* scr, int ncol, int kh) {
  const __bf16* p = scr + ncol * 32 + kh * 16;
  ABU u;
  u.h[0] = *(const v8bf*)(p);
  u.h[1] = *(const v8bf*)(p + 8);
  return u.v;
}

__device__ __forceinline__ float sigm_(float x) { return 1.f / (1.f + __expf(-x)); }

__global__ void __launch_bounds__(BLOCK, 1)
wavenet_fused(const float* __restrict__ x,
              const float* __restrict__ isw, const float* __restrict__ isb,
              const float* __restrict__ ixw, const float* __restrict__ ixb,
              const float* __restrict__ fw,  const float* __restrict__ fb,
              const float* __restrict__ gw,  const float* __restrict__ gb,
              const float* __restrict__ sw,  const float* __restrict__ sb,
              const float* __restrict__ rw,  const float* __restrict__ rb,
              const float* __restrict__ a1w, const float* __restrict__ a1b,
              const float* __restrict__ a2w, const float* __restrict__ a2b,
              float* __restrict__ out) {
  extern __shared__ char smem[];
  float*  H0  = (float*)(smem + OFF_H0);
  float*  H1  = (float*)(smem + OFF_H1);
  float*  SK  = (float*)(smem + OFF_SK);
  __bf16* sWF = (__bf16*)(smem + OFF_WF);
  __bf16* sWG = (__bf16*)(smem + OFF_WG);
  __bf16* sWS = (__bf16*)(smem + OFF_WS);
  __bf16* sWR = (__bf16*)(smem + OFF_WR);
  float*  sBF = (float*)(smem + OFF_BF);
  float*  sBG = (float*)(smem + OFF_BG);
  float*  sBS = (float*)(smem + OFF_BS);
  float*  sBR = (float*)(smem + OFF_BR);
  __bf16* sA1W = (__bf16*)(smem + OFF_A1W);
  __bf16* sA2W = (__bf16*)(smem + OFF_A2W);
  float*  sA1B = (float*)(smem + OFF_A1B);
  float*  sA2B = (float*)(smem + OFF_A2B);
  __bf16* sTR  = (__bf16*)(smem + OFF_TR);

  const int tid    = threadIdx.x;
  const int wave   = tid >> 5;
  const int lane   = tid & 31;
  const int nlane  = lane & 15;      // column within a 16-wide tile
  const int kh     = lane >> 4;      // operand half select per ISA layouts
  const int chunk0 = blockIdx.x * TS;
  const int bidx   = blockIdx.y;
  __bf16* myscr = sTR + wave * 512;

  // ---- Phase 0: stage head weights (once), init h0/skip from x -------------
  for (int i = tid; i < 32 * 32; i += BLOCK) sA1W[i] = (__bf16)a1w[i];
  for (int i = tid; i < 256 * 32; i += BLOCK) sA2W[i] = (__bf16)a2w[i];
  if (tid < 32) sA1B[tid] = a1b[tid];
  for (int i = tid; i < 256; i += BLOCK) sA2B[i] = a2b[i];

  for (int s = tid; s < PADT; s += BLOCK) {
    int gt = chunk0 + s - HALO;
    float xv = (gt >= 0) ? x[(size_t)bidx * T_LEN + gt] : 0.f;
    float* hp = H0 + s * 32;
    float* sp = SK + s * 32;
#pragma unroll 8
    for (int c = 0; c < 32; ++c) {
      hp[c] = ixw[c] * xv + ixb[c];
      sp[c] = isw[c] * xv + isb[c];
    }
  }
  __syncthreads();

  // ---- Phase 1: 15 gated residual layers -----------------------------------
  float* hsrc = H0;
  float* hdst = H1;
  for (int L = 0; L < NLAYERS; ++L) {
    const int d = 1 << (L % 5);
    // stage this layer's weights as bf16 [out][K];  K<32: tap t-d, K>=32: tap t
    for (int i = tid; i < 2048; i += BLOCK) {
      int o = i >> 6, k = i & 63, c = k & 31, tap = k >> 5;
      int gidx = ((L * 32 + o) * 32 + c) * 2 + tap;
      sWF[i] = (__bf16)fw[gidx];
      sWG[i] = (__bf16)gw[gidx];
    }
    for (int i = tid; i < 1024; i += BLOCK) {
      sWS[i] = (__bf16)sw[L * 1024 + i];
      sWR[i] = (__bf16)rw[L * 1024 + i];
    }
    if (tid < 32) {
      sBF[tid] = fb[L * 32 + tid];
      sBG[tid] = gb[L * 32 + tid];
      sBS[tid] = sb[L * 32 + tid];
      sBR[tid] = rb[L * 32 + tid];
    }
    __syncthreads();

    for (int tile = wave; tile < NT_PAD; tile += WAVES) {
      const int n = tile * 16 + nlane;           // local column this lane owns
      // B operands: tap at t and (causally masked) tap at t-d
      v16bf b_t = load_h_b(hsrc, n, kh);
      int s0 = n - d;
      bool ok = (s0 >= 0) && (chunk0 + s0 - HALO >= 0);
      v16bf b_d = load_h_b(hsrc, s0 < 0 ? 0 : s0, kh);
      if (!ok) {
#pragma unroll
        for (int i = 0; i < 16; ++i) b_d[i] = (__bf16)0.f;
      }
      // filter / gate: D = Wk0*h(t-d) + Wk1*h(t) + bias
      v8f f0 = load_bias_tile(sBF, 0, kh);
      v8f f1 = load_bias_tile(sBF, 16, kh);
      v8f g0 = load_bias_tile(sBG, 0, kh);
      v8f g1 = load_bias_tile(sBG, 16, kh);
      f0 = wmma_bf16(load_a(sWF, 128, 0, 0, nlane, kh), b_d, f0);
      f0 = wmma_bf16(load_a(sWF, 128, 0, 64, nlane, kh), b_t, f0);
      f1 = wmma_bf16(load_a(sWF, 128, 16, 0, nlane, kh), b_d, f1);
      f1 = wmma_bf16(load_a(sWF, 128, 16, 64, nlane, kh), b_t, f1);
      g0 = wmma_bf16(load_a(sWG, 128, 0, 0, nlane, kh), b_d, g0);
      g0 = wmma_bf16(load_a(sWG, 128, 0, 64, nlane, kh), b_t, g0);
      g1 = wmma_bf16(load_a(sWG, 128, 16, 0, nlane, kh), b_d, g1);
      g1 = wmma_bf16(load_a(sWG, 128, 16, 64, nlane, kh), b_t, g1);
      // gated activation
      v8f o0, o1;
#pragma unroll
      for (int i = 0; i < 8; ++i) {
        o0[i] = tanhf(f0[i]) * sigm_(g0[i]);
        o1[i] = tanhf(f1[i]) * sigm_(g1[i]);
      }
      // relayout D -> B operand via per-wave LDS pad
      store_tr(myscr, nlane, 0, kh, o0);
      store_tr(myscr, nlane, 1, kh, o1);
      asm volatile("s_wait_dscnt 0" ::: "memory");
      v16bf b_o = load_tr_b(myscr, nlane, kh);
      // skip contribution, accumulate in f32 LDS
      v8f sc0 = wmma_bf16(load_a(sWS, 64, 0, 0, nlane, kh), b_o,
                          load_bias_tile(sBS, 0, kh));
      v8f sc1 = wmma_bf16(load_a(sWS, 64, 16, 0, nlane, kh), b_o,
                          load_bias_tile(sBS, 16, kh));
      {
        v4f* p0 = (v4f*)(SK + n * 32 + kh * 8);
        v4f* p1 = (v4f*)(SK + n * 32 + 16 + kh * 8);
        v4f a0 = p0[0], a1v = p0[1], a2 = p1[0], a3 = p1[1];
#pragma unroll
        for (int i = 0; i < 4; ++i) {
          a0[i] += sc0[i]; a1v[i] += sc0[4 + i];
          a2[i] += sc1[i]; a3[i] += sc1[4 + i];
        }
        p0[0] = a0; p0[1] = a1v; p1[0] = a2; p1[1] = a3;
      }
      // residual: h_next = Wr*out + br + h_prev
      v8f rc0 = wmma_bf16(load_a(sWR, 64, 0, 0, nlane, kh), b_o,
                          load_bias_tile(sBR, 0, kh));
      v8f rc1 = wmma_bf16(load_a(sWR, 64, 16, 0, nlane, kh), b_o,
                          load_bias_tile(sBR, 16, kh));
      {
        const v4f* hp0 = (const v4f*)(hsrc + n * 32 + kh * 8);
        const v4f* hp1 = (const v4f*)(hsrc + n * 32 + 16 + kh * 8);
        v4f* hn0 = (v4f*)(hdst + n * 32 + kh * 8);
        v4f* hn1 = (v4f*)(hdst + n * 32 + 16 + kh * 8);
        v4f a0 = hp0[0], a1v = hp0[1], a2 = hp1[0], a3 = hp1[1];
#pragma unroll
        for (int i = 0; i < 4; ++i) {
          a0[i] += rc0[i]; a1v[i] += rc0[4 + i];
          a2[i] += rc1[i]; a3[i] += rc1[4 + i];
        }
        hn0[0] = a0; hn0[1] = a1v; hn1[0] = a2; hn1[1] = a3;
      }
    }
    __syncthreads();
    float* tmp = hsrc; hsrc = hdst; hdst = tmp;
  }

  // ---- Phase 2: head (relu -> 32x32 -> relu -> 256x32 -> softmax) ----------
  for (int tile = wave; tile < NT_OUT; tile += WAVES) {
    const int n = HALO + tile * 16 + nlane;
    v16bf b_y = load_skip_relu_b(SK, n, kh);
    v8f a0 = wmma_bf16(load_a(sA1W, 64, 0, 0, nlane, kh), b_y,
                       load_bias_tile(sA1B, 0, kh));
    v8f a1t = wmma_bf16(load_a(sA1W, 64, 16, 0, nlane, kh), b_y,
                        load_bias_tile(sA1B, 16, kh));
#pragma unroll
    for (int i = 0; i < 8; ++i) {
      a0[i] = fmaxf(a0[i], 0.f);
      a1t[i] = fmaxf(a1t[i], 0.f);
    }
    store_tr(myscr, nlane, 0, kh, a0);
    store_tr(myscr, nlane, 1, kh, a1t);
    asm volatile("s_wait_dscnt 0" ::: "memory");
    v16bf b_a = load_tr_b(myscr, nlane, kh);

    v8f lg[16];
#pragma unroll
    for (int mt = 0; mt < 16; ++mt)
      lg[mt] = wmma_bf16(load_a(sA2W, 64, mt * 16, 0, nlane, kh), b_a,
                         load_bias_tile(sA2B, mt * 16, kh));

    // softmax over 256 vocab; each column's values split across lane L / L+16
    float mx = -3.402823466e+38f;
#pragma unroll
    for (int mt = 0; mt < 16; ++mt)
#pragma unroll
      for (int i = 0; i < 8; ++i) mx = fmaxf(mx, lg[mt][i]);
    mx = fmaxf(mx, __shfl_xor(mx, 16, 32));
    float sum = 0.f;
#pragma unroll
    for (int mt = 0; mt < 16; ++mt)
#pragma unroll
      for (int i = 0; i < 8; ++i) {
        float e = __expf(lg[mt][i] - mx);
        lg[mt][i] = e;
        sum += e;
      }
    sum += __shfl_xor(sum, 16, 32);
    const float inv = 1.f / sum;

    const int gt = chunk0 + tile * 16 + nlane;
    float* ob = out + (size_t)bidx * 256 * T_LEN + gt;
#pragma unroll
    for (int mt = 0; mt < 16; ++mt)
#pragma unroll
      for (int i = 0; i < 8; ++i) {
        int M = mt * 16 + kh * 8 + i;
        ob[(size_t)M * T_LEN] = lg[mt][i] * inv;
      }
  }
}

extern "C" void kernel_launch(void* const* d_in, const int* in_sizes, int n_in,
                              void* d_out, int out_size, void* d_ws, size_t ws_size,
                              hipStream_t stream) {
  (void)in_sizes; (void)n_in; (void)out_size; (void)d_ws; (void)ws_size;
  const float* x   = (const float*)d_in[0];
  const float* isw = (const float*)d_in[1];
  const float* isb = (const float*)d_in[2];
  const float* ixw = (const float*)d_in[3];
  const float* ixb = (const float*)d_in[4];
  const float* fw  = (const float*)d_in[5];
  const float* fb  = (const float*)d_in[6];
  const float* gw  = (const float*)d_in[7];
  const float* gb  = (const float*)d_in[8];
  const float* sw  = (const float*)d_in[9];
  const float* sb  = (const float*)d_in[10];
  const float* rw  = (const float*)d_in[11];
  const float* rb  = (const float*)d_in[12];
  const float* a1w = (const float*)d_in[13];
  const float* a1b = (const float*)d_in[14];
  const float* a2w = (const float*)d_in[15];
  const float* a2b = (const float*)d_in[16];
  float* out = (float*)d_out;

  (void)hipFuncSetAttribute((const void*)wavenet_fused,
                            hipFuncAttributeMaxDynamicSharedMemorySize,
                            (int)SMEM_BYTES);
  dim3 grid(T_LEN / TS, BATCH_);
  wavenet_fused<<<grid, BLOCK, SMEM_BYTES, stream>>>(
      x, isw, isb, ixw, ixb, fw, fb, gw, gb, sw, sb, rw, rb,
      a1w, a1b, a2w, a2b, out);
}
